// ResNeXt_mpa_32993938768076
// MI455X (gfx1250) — matrix operand order repaired and checked by reading, not compiled, not run on hardware
//
#include <hip/hip_runtime.h>
#include <hip/hip_bf16.h>
#include <cstdint>

// ---------------------------------------------------------------------------
// CDNA5 / gfx1250 wave32 WMMA implementation.
// D = 16x16 f32 accum, A/B = 16x32 bf16 via v_wmma_f32_16x16x32_bf16.
// bf16 A-tiles are moved by the Tensor Data Mover (tensor_load_to_lds +
// s_wait_tensorcnt) with TDM row padding reproducing the 80B LDS row stride.
// f32->bf16 packing uses native __bf16 truncation (fptrunc, RNE) so the
// backend can select v_cvt_pk_bf16_f32.
// ---------------------------------------------------------------------------

typedef __attribute__((ext_vector_type(16))) __bf16        v16bf;
typedef __attribute__((ext_vector_type(4)))  __bf16        v4bf;
typedef __attribute__((ext_vector_type(8)))  float         v8f;
typedef __attribute__((ext_vector_type(4)))  unsigned int  u32x4;
typedef __attribute__((ext_vector_type(4)))  float         f32x4;
typedef __attribute__((ext_vector_type(8)))  int           i32x8;
typedef __attribute__((ext_vector_type(4)))  int           i32x4;

#define BM 128
#define BN 128
#define BK 32
#define LDSS 40   // shorts per LDS row: 32 data + 8 pad -> 80B rows (16B aligned)

__device__ __forceinline__ unsigned short f2bf16(float f) {
  __bf16 h = (__bf16)f;                       // fptrunc f32->bf16, RNE
  return __builtin_bit_cast(unsigned short, h);
}

__device__ __forceinline__ u32x4 pack8(f32x4 a, f32x4 b, float s) {
  struct { v4bf lo, hi; } r;
  r.lo = __builtin_convertvector(a * s, v4bf);  // -> v_cvt_pk_bf16_f32 pairs
  r.hi = __builtin_convertvector(b * s, v4bf);
  return __builtin_bit_cast(u32x4, r);
}

struct Pair { u32x4 a, b; };
__device__ __forceinline__ v16bf frag_from(const unsigned short* p0,
                                           const unsigned short* p1) {
  Pair r;
  r.a = *(const u32x4*)p0;
  r.b = *(const u32x4*)p1;
  return __builtin_bit_cast(v16bf, r);
}

// ---------------------------------------------------------------------------
// TDM: DMA a 128x32 bf16 tile (row stride lda elements) from global memory
// into LDS at byte offset ldsOff, padding each 64B row to 80B so the LDS
// layout matches As[row*LDSS]. OOB columns/rows (beyond valid_cols/rows,
// measured from the tile origin) are written as zero by the TDM.
// D# layout per CDNA5 ISA.
// ---------------------------------------------------------------------------
__device__ __forceinline__ void tdm_load_tile_bf16(const unsigned short* gsrc,
                                                   unsigned ldsOff, int lda,
                                                   int valid_cols, int valid_rows) {
  unsigned long long ga = (unsigned long long)(uintptr_t)gsrc;
  u32x4 g0;
  g0.x = 1u;                                          // count=1, is_restore=0, no gather
  g0.y = ldsOff;                                      // lds_addr (bytes)
  g0.z = (unsigned)(ga & 0xFFFFFFFFull);              // global_addr[31:0]
  g0.w = (unsigned)((ga >> 32) & 0x01FFFFFFull)       // global_addr[56:32]
       | (2u << 30);                                  // type=2 ("image")
  i32x8 g1;
  g1[0] = (1 << 16)     // data_size: 2 bytes
        | (1 << 20)     // pad_enable
        | (3 << 22)     // pad_interval: 16 DWORDs (one 32-elem bf16 row)
        | (3 << 25);    // pad_amount: 4 DWORDs (8 bf16) -> 80B LDS row stride
  g1[1] = (valid_cols & 0xFFFF) << 16;                               // tensor_dim0[15:0]
  g1[2] = ((unsigned)valid_cols >> 16) | ((valid_rows & 0xFFFF) << 16); // dim0 hi | dim1 lo
  g1[3] = ((unsigned)valid_rows >> 16) | (BK << 16);                 // dim1 hi | tile_dim0=32
  g1[4] = BM;                                                        // tile_dim1=128, tile_dim2=0
  g1[5] = lda;                                                       // tensor_dim0_stride[31:0]
  g1[6] = 0;                                                         // stride hi | dim1_stride lo
  g1[7] = 0;
  i32x4 z4 = {0, 0, 0, 0};
#if defined(__clang_major__) && __clang_major__ >= 23
  i32x8 z8 = {0, 0, 0, 0, 0, 0, 0, 0};
  __builtin_amdgcn_tensor_load_to_lds(g0, g1, z4, z4, z8, 0);
#else
  __builtin_amdgcn_tensor_load_to_lds(g0, g1, z4, z4, 0);
#endif
}

// ---------------------------------------------------------------------------
// Row-norm reciprocal:  rqn[b] = 1/max(||q_b||, eps),  rpn[c] likewise.
// ---------------------------------------------------------------------------
__global__ __launch_bounds__(256)
void rownorm_kernel(const float* __restrict__ q, const float* __restrict__ p,
                    float* __restrict__ rqn, float* __restrict__ rpn,
                    int B, int C, int D) {
  __shared__ float red[256];
  int row = blockIdx.x, t = threadIdx.x;
  const float* src;
  float* dst;
  if (row < B) { src = q + (size_t)row * D; dst = rqn + row; }
  else         { src = p + (size_t)(row - B) * D; dst = rpn + (row - B); }
  float s = 0.f;
  for (int i = t; i < D; i += 256) { float x = src[i]; s += x * x; }
  red[t] = s; __syncthreads();
  for (int st = 128; st > 0; st >>= 1) {
    if (t < st) red[t] += red[t + st];
    __syncthreads();
  }
  if (t == 0) {
    float n = sqrtf(red[0]);
    *dst = 1.f / fmaxf(n, 1e-8f);
  }
}

// ---------------------------------------------------------------------------
// Softmax over cos row (len C), temperature 0.2 (x5), with the
// "sum(cos)==0 -> zero row" mask folded in. attn stored bf16, stride ld.
// ---------------------------------------------------------------------------
__global__ __launch_bounds__(256)
void softmax_kernel(const float* __restrict__ S, unsigned short* __restrict__ attn,
                    int C, int ld) {
  __shared__ float red[256];
  const float INV_T = 5.0f;
  int row = blockIdx.x, t = threadIdx.x;
  const float* Sr = S + (size_t)row * C;
  float v[4];
  float lsum = 0.f, lmax = -3.4e38f;
#pragma unroll
  for (int i = 0; i < 4; i++) {
    int c = t + i * 256;
    v[i] = (c < C) ? Sr[c] : 0.f;
    if (c < C) { lsum += v[i]; lmax = fmaxf(lmax, v[i]); }
  }
  red[t] = lsum; __syncthreads();
  for (int st = 128; st > 0; st >>= 1) { if (t < st) red[t] += red[t + st]; __syncthreads(); }
  float rawsum = red[0]; __syncthreads();
  red[t] = lmax; __syncthreads();
  for (int st = 128; st > 0; st >>= 1) { if (t < st) red[t] = fmaxf(red[t], red[t + st]); __syncthreads(); }
  float mx = red[0]; __syncthreads();
  float e[4]; float lexp = 0.f;
#pragma unroll
  for (int i = 0; i < 4; i++) {
    int c = t + i * 256;
    e[i] = (c < C) ? __expf((v[i] - mx) * INV_T) : 0.f;
    lexp += e[i];
  }
  red[t] = lexp; __syncthreads();
  for (int st = 128; st > 0; st >>= 1) { if (t < st) red[t] += red[t + st]; __syncthreads(); }
  float denom = red[0];
  float scale = (rawsum != 0.f) ? (1.f / denom) : 0.f;  // mask fused here
#pragma unroll
  for (int i = 0; i < 4; i++) {
    int c = t + i * 256;
    if (c < C) attn[(size_t)row * ld + c] = f2bf16(e[i] * scale);
  }
}

// ---------------------------------------------------------------------------
// Tiled bf16 WMMA GEMM:  C[M,N] = A[M,K] . B^T  (B logically [N,K]).
// Block tile 128x128x32, 8 waves (4 in M x 2 in N), wave tile 32x64
// = 2x4 v_wmma_f32_16x16x32_bf16 accumulators.
//
// MODE 0 (cos):      A = querys f32 (scaled by rqn on fill),
//                    B = prototype f32 NT (scaled by rpn on fill),
//                    epilogue: Cout = scores f32.
// MODE 1 (enhanced): A = attn bf16 via TDM, B = prototype f32 NN
//                    (transposed into LDS on fill), epilogue:
//                    x = querys + acc, stored bf16 into Cbf.
// MODE 2 (logits):   A = x bf16 via TDM, B = fc_w f32 NT, epilogue:
//                    Cout = acc + fc_b.
// ---------------------------------------------------------------------------
template <int MODE>
__global__ __launch_bounds__(256)
void gemm_nt_wmma(const float* __restrict__ Af32, const unsigned short* __restrict__ Abf,
                  int lda,
                  const float* __restrict__ Bf32, int ldb,
                  const float* __restrict__ ascale, const float* __restrict__ bscale,
                  const float* __restrict__ q_add, const float* __restrict__ fcb,
                  float* __restrict__ Cout, unsigned short* __restrict__ Cbf,
                  int M, int N, int K) {
  __shared__ unsigned short As[BM * LDSS];
  __shared__ unsigned short Bs[BN * LDSS];

  const int t    = threadIdx.x;
  const int lane = t & 31;
  const int wave = t >> 5;
  const int wm   = wave & 3;   // wave row: 0..3 -> M offset wm*32
  const int wn   = wave >> 2;  // wave col: 0..1 -> N offset wn*64
  const int half = lane >> 4;
  const int l15  = lane & 15;

  const int blockM = blockIdx.y * BM;
  const int blockN = blockIdx.x * BN;

  // LDS byte offset of As (flat LDS address truncates to the low 32 bits)
  const unsigned asOff = (unsigned)(uintptr_t)(&As[0]);

  v8f acc[2][4] = {};

  for (int k0 = 0; k0 < K; k0 += BK) {
    __syncthreads();

    // ---- fill A tile ----
    if (MODE == 0) {
      // f32 source scaled by 1/||q||; unit = 8 bf16 (16B)
#pragma unroll
      for (int uu = 0; uu < 2; uu++) {
        int u   = t + uu * 256;          // 0..511
        int row = u >> 2;
        int kc  = (u & 3) * 8;
        int gm  = blockM + row;
        int gk  = k0 + kc;               // K=2048: always in range
        const f32x4* src = (const f32x4*)(Af32 + (size_t)gm * lda + gk);
        *(u32x4*)&As[row * LDSS + kc] = pack8(src[0], src[1], ascale[gm]);
      }
    } else {
      // bf16 source: Tensor Data Mover, issued by wave 0 only.
      if (wave == 0) {
        tdm_load_tile_bf16(Abf + (size_t)blockM * lda + k0, asOff, lda,
                           K - k0, M - blockM);
      }
    }

    // ---- fill B tile: LDS layout [n][k] (overlaps with the TDM transfer) ----
    if (MODE == 1) {
      // NN source: B stored [K,N] row-major -> transpose on fill.
#pragma unroll
      for (int i = 0; i < 16; i++) {
        int e  = t + i * 256;            // 0..4095
        int nl = e & 127;
        int kc = e >> 7;
        int gn = blockN + nl;            // N=2048, grid exact
        int gk = k0 + kc;
        float vv = (gk < K) ? Bf32[(size_t)gk * ldb + gn] : 0.f;
        Bs[nl * LDSS + kc] = f2bf16(vv);
      }
    } else {
#pragma unroll
      for (int uu = 0; uu < 2; uu++) {
        int u  = t + uu * 256;
        int nl = u >> 2;
        int kc = (u & 3) * 8;
        int gn = blockN + nl;
        int gk = k0 + kc;                // K=2048: always in range
        u32x4 val = {0u, 0u, 0u, 0u};
        if (gn < N) {
          const f32x4* src = (const f32x4*)(Bf32 + (size_t)gn * ldb + gk);
          float s = (MODE == 0) ? bscale[gn] : 1.0f;
          val = pack8(src[0], src[1], s);
        }
        *(u32x4*)&Bs[nl * LDSS + kc] = val;
      }
    }

    // ---- prefetch next k-tile of the B stream (global_prefetch_b8) ----
    if (k0 + BK < K && t < 128) {
      if (MODE == 1) {
        // next tile rows k0+32..k0+63, 512B per row -> 4 lines per row
        int gk = k0 + BK + (t >> 2);
        if (gk < K)
          __builtin_prefetch(Bf32 + (size_t)gk * ldb + blockN + (t & 3) * 32, 0, 1);
      } else {
        // next tile: 128 rows x 128B (32 f32) -> one line per row
        int gn = blockN + t;
        if (gn < N)
          __builtin_prefetch(Bf32 + (size_t)gn * ldb + k0 + BK, 0, 1);
      }
    }

    if (MODE != 0) {
      if (wave == 0) __builtin_amdgcn_s_wait_tensorcnt(0);  // TDM done (wave0)
    }
    __syncthreads();

    // ---- fragments per ISA layout ----
    // A 16x32 bf16: lane<16 -> row M=l15, K {0..7,16..23}; lane>=16 -> K {8..15,24..31}
    v16bf af[2];
#pragma unroll
    for (int mi = 0; mi < 2; mi++) {
      const unsigned short* p = &As[(wm * 32 + mi * 16 + l15) * LDSS + half * 8];
      af[mi] = frag_from(p, p + 16);
    }
    // B 32x16 bf16 (stored [n][k]): lane holds column n=l15; K {0..7,16..23} / {8..15,24..31} by half
    v16bf bfg[4];
#pragma unroll
    for (int ni = 0; ni < 4; ni++) {
      const unsigned short* p = &Bs[(wn * 64 + ni * 16 + l15) * LDSS + half * 8];
      bfg[ni] = frag_from(p, p + 16);
    }

#pragma unroll
    for (int mi = 0; mi < 2; mi++)
#pragma unroll
      for (int ni = 0; ni < 4; ni++)
        acc[mi][ni] = __builtin_amdgcn_wmma_f32_16x16x32_bf16(
            false, af[mi], false, bfg[ni], (short)0, acc[mi][ni], false, false);
  }

  // ---- epilogue: C/D layout -> (VGPR j, lane) = C[half*8+j][l15] ----
#pragma unroll
  for (int mi = 0; mi < 2; mi++) {
#pragma unroll
    for (int ni = 0; ni < 4; ni++) {
#pragma unroll
      for (int j = 0; j < 8; j++) {
        int gm = blockM + wm * 32 + mi * 16 + half * 8 + j;
        int gn = blockN + wn * 64 + ni * 16 + l15;
        float v = acc[mi][ni][j];
        if (MODE == 0) {
          if (gn < N) Cout[(size_t)gm * N + gn] = v;
        } else if (MODE == 1) {
          float xv = q_add[(size_t)gm * N + gn] + v;  // x = q + enhanced
          Cbf[(size_t)gm * N + gn] = f2bf16(xv);
        } else {
          if (gn < N) Cout[(size_t)gm * N + gn] = v + fcb[gn];
        }
      }
    }
  }
}

// ---------------------------------------------------------------------------
extern "C" void kernel_launch(void* const* d_in, const int* in_sizes, int n_in,
                              void* d_out, int out_size, void* d_ws, size_t ws_size,
                              hipStream_t stream) {
  const float* querys    = (const float*)d_in[0];  // [B,D]
  const float* prototype = (const float*)d_in[1];  // [C,D]
  const float* fc_w      = (const float*)d_in[2];  // [C,D]
  const float* fc_b      = (const float*)d_in[3];  // [C]

  const int B = 16384, C = 1000, D = 2048;
  const int LDA_ATTN = 1024;  // padded stride for 16B-aligned bf16 rows

  char* ws = (char*)d_ws;
  float* rqn = (float*)ws;              ws += (size_t)B * sizeof(float);
  float* rpn = (float*)ws;              ws += 4096;
  float* S   = (float*)ws;              ws += (size_t)B * C * sizeof(float);
  unsigned short* attn = (unsigned short*)ws; ws += (size_t)B * LDA_ATTN * 2;
  unsigned short* xbf  = (unsigned short*)ws; ws += (size_t)B * D * 2;

  float* out = (float*)d_out;           // [B*C logits | B*D embedding]

  // 1) reciprocal row norms
  rownorm_kernel<<<B + C, 256, 0, stream>>>(querys, prototype, rqn, rpn, B, C, D);

  // 2) cos = (q/|q|) . (p/|p|)^T   [B,C]
  gemm_nt_wmma<0><<<dim3((C + BN - 1) / BN, B / BM), 256, 0, stream>>>(
      querys, nullptr, D, prototype, D, rqn, rpn, nullptr, nullptr,
      S, nullptr, B, C, D);

  // 3) attn = softmax(cos*5), with sum(cos)==0 mask folded in
  softmax_kernel<<<B, 256, 0, stream>>>(S, attn, C, LDA_ATTN);

  // 4) x = q + attn . prototype    [B,D], stored bf16
  gemm_nt_wmma<1><<<dim3(D / BN, B / BM), 256, 0, stream>>>(
      nullptr, attn, LDA_ATTN, prototype, D, nullptr, nullptr, querys, nullptr,
      nullptr, xbf, B, D, C);

  // 5) logits = x . fc_w^T + fc_b  [B,C] -> d_out[0 : B*C]
  gemm_nt_wmma<2><<<dim3((C + BN - 1) / BN, B / BM), 256, 0, stream>>>(
      nullptr, xbf, D, fc_w, D, nullptr, nullptr, nullptr, fc_b,
      out, nullptr, B, C, D);

  // 6) embedding = querys -> d_out[B*C : B*C + B*D]
  hipMemcpyAsync(out + (size_t)B * C, querys, (size_t)B * D * sizeof(float),
                 hipMemcpyDeviceToDevice, stream);
}
